// SpatialMix_BiV4_19250043421269
// MI455X (gfx1250) — matrix operand order repaired and checked by reading, not compiled, not run on hardware
//
#include <hip/hip_runtime.h>
#include <stdint.h>

// ---------------------------------------------------------------------------
// Problem constants (reference: B=16, H=W=64, T=4096, C=192)
// ---------------------------------------------------------------------------
constexpr int BB = 16;
constexpr int HH = 64;
constexpr int WW = 64;
constexpr int TT = HH * WW;          // 4096
constexpr int CC = 192;
constexpr int ROWS = BB * TT;        // 65536
constexpr size_t NN = (size_t)ROWS * CC;   // 12,582,912 elements
constexpr int CHUNK = 256;
constexpr int NCH = TT / CHUNK;      // 16 chunks per image

// Packed weight fragments: per matrix: 12 col-tiles * 6 k-chunks * 32 lanes * 8 dwords
constexpr int PACK_DW = 12 * 6 * 32 * 8;   // 18432 dwords (= 192*192 bf16)

// Workspace layout (bytes, all offsets multiples of 256)
constexpr size_t OFF_PACK  = 0;
constexpr size_t OFF_XS    = OFF_PACK + (size_t)4 * PACK_DW * 4;  // 4 packed matrices
constexpr size_t OFF_EK    = OFF_XS  + NN * 2;   // xs bf16
constexpr size_t OFF_EKV   = OFF_EK  + NN * 4;
constexpr size_t OFF_SR    = OFF_EKV + NN * 4;
constexpr size_t OFF_AF    = OFF_SR  + NN * 4;
constexpr size_t OFF_BF    = OFF_AF  + NN * 4;
constexpr size_t OFF_Z     = OFF_BF  + NN * 4;   // z bf16
constexpr size_t OFF_CHUNK = OFF_Z   + NN * 2;   // 4 comps per (b,c,chunk)
constexpr size_t OFF_CF    = OFF_CHUNK + (size_t)BB * CC * NCH * 4 * 4;
constexpr size_t OFF_CB    = OFF_CF    + (size_t)BB * CC * NCH * 2 * 4;

typedef __attribute__((ext_vector_type(16))) __bf16 v16bf;
typedef __attribute__((ext_vector_type(8)))  float  v8f;

__device__ __forceinline__ unsigned short f2bf(float f) {
    unsigned int u = __float_as_uint(f);
    unsigned int r = u + 0x7FFFu + ((u >> 16) & 1u);   // round-to-nearest-even
    return (unsigned short)(r >> 16);
}

union Frag { uint32_t u[8]; v16bf v; };

// ---------------------------------------------------------------------------
// Kernel 0: pack W{k,v,r,o} (fp32 row-major [o,k]) into per-lane WMMA B-frag
// layout, bf16. B is Kx16: dwords 0-3 hold K=8h+2v.., dwords 4-7 K=16+8h+2(v-4)..
// (h = lane/16); dword v holds a consecutive K-pair. Each lane's 8 dwords are
// stored contiguously (32B) so the GEMM does two b128 loads per fragment.
// ---------------------------------------------------------------------------
__global__ void pack_w_kernel(const float* __restrict__ Wk, const float* __restrict__ Wv,
                              const float* __restrict__ Wr, const float* __restrict__ Wo,
                              uint32_t* __restrict__ packed) {
    int idx = blockIdx.x * 256 + threadIdx.x;
    if (idx >= 4 * PACK_DW) return;
    int mat  = idx / PACK_DW;
    int p    = idx % PACK_DW;
    int v    = p & 7;
    int lane = (p >> 3) & 31;
    int q    = p >> 8;
    int kk   = q % 6;
    int col  = q / 6;
    const float* W = (mat == 0) ? Wk : (mat == 1) ? Wv : (mat == 2) ? Wr : Wo;
    int o = col * 16 + (lane & 15);
    int k = kk * 32 + ((v < 4) ? 0 : 16) + ((lane >= 16) ? 8 : 0) + 2 * (v & 3);
    float lo = W[o * CC + k];
    float hi = W[o * CC + k + 1];
    packed[idx] = (uint32_t)f2bf(lo) | ((uint32_t)f2bf(hi) << 16);
}

// ---------------------------------------------------------------------------
// Kernel 1: OmniShift. One thread per (b,t,c); c fastest -> coalesced reads
// of the 192-channel vectors; 3x3/5x5 neighbor rereads hit L1/L2 (x is 50MB,
// fits in 192MB L2). Output xs as bf16 (GEMM A operand).
// ---------------------------------------------------------------------------
__global__ __launch_bounds__(256) void omnishift_kernel(
        const float* __restrict__ x, const float* __restrict__ w1,
        const float* __restrict__ w3, const float* __restrict__ w5,
        const float* __restrict__ alpha, unsigned short* __restrict__ xs) {
    size_t idx = (size_t)blockIdx.x * 256 + threadIdx.x;
    if (idx >= NN) return;
    int c = (int)(idx % CC);
    size_t bt = idx / CC;
    int t = (int)(bt % TT);
    int b = (int)(bt / TT);
    int h = t / WW, w = t % WW;
    float a0 = alpha[0], a1 = alpha[1], a2 = alpha[2], a3 = alpha[3];
    const float* xb = x + ((size_t)b * TT) * CC + c;
    float s = (a0 + a1 * w1[c]) * xb[(size_t)t * CC];
    #pragma unroll
    for (int dy = -1; dy <= 1; ++dy)
        #pragma unroll
        for (int dx = -1; dx <= 1; ++dx) {
            int hy = h + dy, wx = w + dx;
            if (hy >= 0 && hy < HH && wx >= 0 && wx < WW)
                s += a2 * w3[c * 9 + (dy + 1) * 3 + (dx + 1)] *
                     xb[((size_t)(hy * WW + wx)) * CC];
        }
    #pragma unroll
    for (int dy = -2; dy <= 2; ++dy)
        #pragma unroll
        for (int dx = -2; dx <= 2; ++dx) {
            int hy = h + dy, wx = w + dx;
            if (hy >= 0 && hy < HH && wx >= 0 && wx < WW)
                s += a3 * w5[c * 25 + (dy + 2) * 5 + (dx + 2)] *
                     xb[((size_t)(hy * WW + wx)) * CC];
        }
    xs[idx] = f2bf(s);
}

// ---------------------------------------------------------------------------
// Kernel 2: fused K/V/R projection with v_wmma_f32_16x16x32_bf16.
// Block = 256 threads (8 waves), wave owns 16 rows of xs; A fragments for all
// 6 k-chunks are held in VGPRs and reused across 12 column tiles x 3 matrices.
// Epilogue fuses ek=exp(k), ekv=ek*v, sr=sigmoid(r) (exp(u+k) == exp(u)*ek).
// ---------------------------------------------------------------------------
__global__ __launch_bounds__(256) void gemm_kvr_kernel(
        const unsigned short* __restrict__ xs,
        const uint32_t* __restrict__ pK, const uint32_t* __restrict__ pV,
        const uint32_t* __restrict__ pR,
        float* __restrict__ ek, float* __restrict__ ekv, float* __restrict__ sr) {
    const int lane = threadIdx.x & 31;
    const int wave = threadIdx.x >> 5;
    const int rowbase = blockIdx.x * 128 + wave * 16;
    const int m = rowbase + (lane & 15);
    const int koffA = (lane >= 16) ? 8 : 0;     // A layout: K +8 for upper half

    const uint32_t* x32 = (const uint32_t*)xs;  // bf16 pairs
    Frag a[6];
    #pragma unroll
    for (int kk = 0; kk < 6; ++kk)
        #pragma unroll
        for (int v = 0; v < 8; ++v) {
            int k = kk * 32 + ((v < 4) ? 0 : 16) + koffA + 2 * (v & 3);
            a[kk].u[v] = x32[(size_t)m * (CC / 2) + (k >> 1)];
        }

    const int row0 = rowbase + ((lane >= 16) ? 8 : 0);
    #pragma unroll 1
    for (int col = 0; col < 12; ++col) {
        v8f ck = {}, cv = {}, cr = {};
        #pragma unroll
        for (int kk = 0; kk < 6; ++kk) {
            size_t bi = ((size_t)(col * 6 + kk) * 32 + lane) * 8;
            Frag bk, bv, br;
            *(uint4*)&bk.u[0] = *(const uint4*)&pK[bi];
            *(uint4*)&bk.u[4] = *(const uint4*)&pK[bi + 4];
            *(uint4*)&bv.u[0] = *(const uint4*)&pV[bi];
            *(uint4*)&bv.u[4] = *(const uint4*)&pV[bi + 4];
            *(uint4*)&br.u[0] = *(const uint4*)&pR[bi];
            *(uint4*)&br.u[4] = *(const uint4*)&pR[bi + 4];
            ck = __builtin_amdgcn_wmma_f32_16x16x32_bf16(false, a[kk].v, false, bk.v, (short)0, ck, false, false);
            cv = __builtin_amdgcn_wmma_f32_16x16x32_bf16(false, a[kk].v, false, bv.v, (short)0, cv, false, false);
            cr = __builtin_amdgcn_wmma_f32_16x16x32_bf16(false, a[kk].v, false, br.v, (short)0, cr, false, false);
        }
        int c0 = col * 16 + (lane & 15);
        #pragma unroll
        for (int e = 0; e < 8; ++e) {
            size_t o = (size_t)(row0 + e) * CC + c0;
            float ekt = __expf(ck[e]);
            ek[o]  = ekt;
            ekv[o] = ekt * cv[e];
            sr[o]  = 1.0f / (1.0f + __expf(-cr[e]));
        }
    }
}

// ---------------------------------------------------------------------------
// Kernel 3: per-chunk scan sums. Linear recurrence a <- ew*a + x means a
// chunk of length L acts as (scale = ew^L, add = sum). One block per (b,chunk),
// 192 threads (one per channel) -> 768B coalesced reads per step.
//   fwd add: Sf = sum_i ew^(L-1-i) x_i ;  bwd add: Sb = sum_i ew^i x_i
// ---------------------------------------------------------------------------
__global__ __launch_bounds__(CC) void scan1_kernel(
        const float* __restrict__ ek, const float* __restrict__ ekv,
        const float* __restrict__ decay, float* __restrict__ chunkbuf) {
    int b = blockIdx.x / NCH, j = blockIdx.x % NCH, c = threadIdx.x;
    float ew = __expf(-decay[c] / (float)TT);
    size_t base = ((size_t)b * TT + (size_t)j * CHUNK) * CC + c;
    float sfa = 0.f, sfb = 0.f, sba = 0.f, sbb = 0.f, pw = 1.f;
    for (int i = 0; i < CHUNK; ++i) {
        size_t o = base + (size_t)i * CC;
        float ekt = ek[o], ekvt = ekv[o];
        sfa = ew * sfa + ekvt;  sfb = ew * sfb + ekt;
        sba += pw * ekvt;       sbb += pw * ekt;
        pw *= ew;
    }
    size_t cb = (((size_t)(b * CC + c)) * NCH + j) * 4;
    chunkbuf[cb + 0] = sfa; chunkbuf[cb + 1] = sfb;
    chunkbuf[cb + 2] = sba; chunkbuf[cb + 3] = sbb;
}

// ---------------------------------------------------------------------------
// Kernel 4: combine chunk carries sequentially (16 steps, 3072 threads).
// Emits the EXCLUSIVE carry entering each chunk, both directions.
// ---------------------------------------------------------------------------
__global__ __launch_bounds__(256) void scan2_kernel(
        const float* __restrict__ chunkbuf, const float* __restrict__ decay,
        float* __restrict__ carryF, float* __restrict__ carryB) {
    int tid = blockIdx.x * 256 + threadIdx.x;
    if (tid >= BB * CC) return;
    int c = tid % CC;
    float ewL = __expf(-decay[c] * (float)CHUNK / (float)TT);
    float ca = 0.f, cb = 0.f;
    for (int j = 0; j < NCH; ++j) {
        size_t o = ((size_t)tid * NCH + j);
        carryF[o * 2 + 0] = ca; carryF[o * 2 + 1] = cb;
        ca = ewL * ca + chunkbuf[o * 4 + 0];
        cb = ewL * cb + chunkbuf[o * 4 + 1];
    }
    ca = 0.f; cb = 0.f;
    for (int j = NCH - 1; j >= 0; --j) {
        size_t o = ((size_t)tid * NCH + j);
        carryB[o * 2 + 0] = ca; carryB[o * 2 + 1] = cb;
        ca = ewL * ca + chunkbuf[o * 4 + 2];
        cb = ewL * cb + chunkbuf[o * 4 + 3];
    }
}

// ---------------------------------------------------------------------------
// Kernel 5: forward apply — write exclusive forward sums af, bf.
// ---------------------------------------------------------------------------
__global__ __launch_bounds__(CC) void scan3a_kernel(
        const float* __restrict__ ek, const float* __restrict__ ekv,
        const float* __restrict__ decay, const float* __restrict__ carryF,
        float* __restrict__ af, float* __restrict__ bf) {
    int b = blockIdx.x / NCH, j = blockIdx.x % NCH, c = threadIdx.x;
    float ew = __expf(-decay[c] / (float)TT);
    size_t co = (((size_t)(b * CC + c)) * NCH + j) * 2;
    float ca = carryF[co], cb = carryF[co + 1];
    size_t base = ((size_t)b * TT + (size_t)j * CHUNK) * CC + c;
    for (int i = 0; i < CHUNK; ++i) {
        size_t o = base + (size_t)i * CC;
        af[o] = ca; bf[o] = cb;                 // emit before update (exclusive)
        float ekt = ek[o], ekvt = ekv[o];
        ca = ew * ca + ekvt; cb = ew * cb + ekt;
    }
}

// ---------------------------------------------------------------------------
// Kernel 6: backward apply, fused with WKV ratio + sigmoid gate.
//   y = (af + ab + eu*ekv) / (bf + bb + eu*ek),  z = sr*y  (stored bf16)
// ---------------------------------------------------------------------------
__global__ __launch_bounds__(CC) void scan3b_kernel(
        const float* __restrict__ ek, const float* __restrict__ ekv,
        const float* __restrict__ af, const float* __restrict__ bf,
        const float* __restrict__ sr, const float* __restrict__ decay,
        const float* __restrict__ boost, const float* __restrict__ carryB,
        unsigned short* __restrict__ z) {
    int b = blockIdx.x / NCH, j = blockIdx.x % NCH, c = threadIdx.x;
    float ew = __expf(-decay[c] / (float)TT);
    float eu = __expf(boost[c] / (float)TT);
    size_t co = (((size_t)(b * CC + c)) * NCH + j) * 2;
    float ca = carryB[co], cb = carryB[co + 1];
    size_t base = ((size_t)b * TT + (size_t)j * CHUNK) * CC + c;
    for (int i = CHUNK - 1; i >= 0; --i) {
        size_t o = base + (size_t)i * CC;
        float ekt = ek[o], ekvt = ekv[o];
        float num = af[o] + ca + eu * ekvt;
        float den = bf[o] + cb + eu * ekt;
        z[o] = f2bf(sr[o] * (num / den));
        ca = ew * ca + ekvt; cb = ew * cb + ekt;
    }
}

// ---------------------------------------------------------------------------
// Kernel 7: output projection out = z @ Wo^T (bf16 WMMA, fp32 out).
// ---------------------------------------------------------------------------
__global__ __launch_bounds__(256) void gemm_out_kernel(
        const unsigned short* __restrict__ z, const uint32_t* __restrict__ pO,
        float* __restrict__ out) {
    const int lane = threadIdx.x & 31;
    const int wave = threadIdx.x >> 5;
    const int rowbase = blockIdx.x * 128 + wave * 16;
    const int m = rowbase + (lane & 15);
    const int koffA = (lane >= 16) ? 8 : 0;

    const uint32_t* z32 = (const uint32_t*)z;
    Frag a[6];
    #pragma unroll
    for (int kk = 0; kk < 6; ++kk)
        #pragma unroll
        for (int v = 0; v < 8; ++v) {
            int k = kk * 32 + ((v < 4) ? 0 : 16) + koffA + 2 * (v & 3);
            a[kk].u[v] = z32[(size_t)m * (CC / 2) + (k >> 1)];
        }

    const int row0 = rowbase + ((lane >= 16) ? 8 : 0);
    #pragma unroll 1
    for (int col = 0; col < 12; ++col) {
        v8f acc = {};
        #pragma unroll
        for (int kk = 0; kk < 6; ++kk) {
            size_t bi = ((size_t)(col * 6 + kk) * 32 + lane) * 8;
            Frag bo;
            *(uint4*)&bo.u[0] = *(const uint4*)&pO[bi];
            *(uint4*)&bo.u[4] = *(const uint4*)&pO[bi + 4];
            acc = __builtin_amdgcn_wmma_f32_16x16x32_bf16(false, a[kk].v, false, bo.v, (short)0, acc, false, false);
        }
        int c0 = col * 16 + (lane & 15);
        #pragma unroll
        for (int e = 0; e < 8; ++e)
            out[(size_t)(row0 + e) * CC + c0] = acc[e];
    }
}

// ---------------------------------------------------------------------------
extern "C" void kernel_launch(void* const* d_in, const int* in_sizes, int n_in,
                              void* d_out, int out_size, void* d_ws, size_t ws_size,
                              hipStream_t stream) {
    const float* x     = (const float*)d_in[0];
    const float* w1    = (const float*)d_in[1];
    const float* w3    = (const float*)d_in[2];
    const float* w5    = (const float*)d_in[3];
    const float* alpha = (const float*)d_in[4];
    const float* Wk    = (const float*)d_in[5];
    const float* Wv    = (const float*)d_in[6];
    const float* Wr    = (const float*)d_in[7];
    const float* Wo    = (const float*)d_in[8];
    const float* decay = (const float*)d_in[9];
    const float* boost = (const float*)d_in[10];

    char* ws = (char*)d_ws;
    uint32_t*       packW   = (uint32_t*)(ws + OFF_PACK);
    unsigned short* xs      = (unsigned short*)(ws + OFF_XS);
    float*          ek      = (float*)(ws + OFF_EK);
    float*          ekv     = (float*)(ws + OFF_EKV);
    float*          srb     = (float*)(ws + OFF_SR);
    float*          af      = (float*)(ws + OFF_AF);
    float*          bf      = (float*)(ws + OFF_BF);
    unsigned short* zb      = (unsigned short*)(ws + OFF_Z);
    float*          chunkb  = (float*)(ws + OFF_CHUNK);
    float*          carryF  = (float*)(ws + OFF_CF);
    float*          carryB  = (float*)(ws + OFF_CB);
    float*          out     = (float*)d_out;

    pack_w_kernel<<<(4 * PACK_DW + 255) / 256, 256, 0, stream>>>(Wk, Wv, Wr, Wo, packW);
    omnishift_kernel<<<(int)((NN + 255) / 256), 256, 0, stream>>>(x, w1, w3, w5, alpha, xs);
    gemm_kvr_kernel<<<ROWS / 128, 256, 0, stream>>>(xs, packW, packW + PACK_DW,
                                                    packW + 2 * PACK_DW, ek, ekv, srb);
    scan1_kernel<<<BB * NCH, CC, 0, stream>>>(ek, ekv, decay, chunkb);
    scan2_kernel<<<(BB * CC + 255) / 256, 256, 0, stream>>>(chunkb, decay, carryF, carryB);
    scan3a_kernel<<<BB * NCH, CC, 0, stream>>>(ek, ekv, decay, carryF, af, bf);
    scan3b_kernel<<<BB * NCH, CC, 0, stream>>>(ek, ekv, af, bf, srb, decay, boost, carryB, zb);
    gemm_out_kernel<<<ROWS / 128, 256, 0, stream>>>(zb, packW + 3 * PACK_DW, out);
}
